// SNN_EMNIST_20186346291688
// MI455X (gfx1250) — matrix operand order repaired and checked by reading, not compiled, not run on hardware
//
#include <hip/hip_runtime.h>

typedef __attribute__((ext_vector_type(16))) _Float16 v16h;
typedef __attribute__((ext_vector_type(8)))  float    v8f;

#define TSTEPS 10
#define IN_DIM 784
#define KPAD   800          // 25 k-tiles of 32
#define HID    256
#define NCLS   47
#define BATCH  4096
#define ROWS   64           // batch rows per block
#define NBLK   (BATCH/ROWS) // 64
#define KT1    25           // ceil(784/32)
#define KT2    8            // 256/32
#define NT1    16           // 256/16
#define NT2    3            // 48/16 (NCLS padded to 48)

// ---------------------------------------------------------------------------
// x -> u16 Bernoulli thresholds, K padded to 800 (pad threshold 0 = no spike).
// spike = (rand_u16 < thr) matches P(u < x) to within 2^-16.
// ---------------------------------------------------------------------------
__global__ void pack_thr_kernel(const float* __restrict__ x, unsigned short* __restrict__ thr)
{
    int gid = blockIdx.x * blockDim.x + threadIdx.x;     // BATCH*KPAD
    if (gid >= BATCH * KPAD) return;
    int row = gid / KPAD;
    int k   = gid - row * KPAD;
    unsigned short t = 0;
    if (k < IN_DIM) {
        float v = x[(size_t)row * IN_DIM + k];
        v = fminf(fmaxf(v, 0.0f), 0.999985f);
        t = (unsigned short)(v * 65536.0f);
    }
    thr[gid] = t;
}

// ---------------------------------------------------------------------------
// Weight pre-packing into WMMA B-fragment layout (16-bit B, 32x16 per tile):
//   lanes 0-15 : K = 0..7, 16..23  (VGPR r -> K = 2r, 2r+1 (+8 for r>=4)), N = lane&15
//   lanes 16-31: K = 8..15, 24..31
// Fragment f stored as 32 lanes x 16 halfs, contiguous per lane.
// ---------------------------------------------------------------------------
__global__ void pack_w1_kernel(const float* __restrict__ W1, _Float16* __restrict__ out)
{
    int gid = blockIdx.x * blockDim.x + threadIdx.x;     // KT1*NT1*32 = 12800
    if (gid >= KT1 * NT1 * 32) return;
    int lane = gid & 31;
    int f    = gid >> 5;
    int nt   = f & 15;
    int kt   = f >> 4;
    int n     = nt * 16 + (lane & 15);
    int kbase = kt * 32 + ((lane & 16) ? 8 : 0);
    _Float16* dst = out + (size_t)gid * 16;
#pragma unroll
    for (int r = 0; r < 8; ++r) {
#pragma unroll
        for (int p = 0; p < 2; ++p) {
            int k = kbase + 2 * r + p + ((r >= 4) ? 8 : 0);
            float v = (k < IN_DIM) ? W1[(size_t)k * HID + n] : 0.0f;
            dst[2 * r + p] = (_Float16)v;
        }
    }
}

__global__ void pack_wr_kernel(const float* __restrict__ Wr, _Float16* __restrict__ out)
{
    int gid = blockIdx.x * blockDim.x + threadIdx.x;     // KT2*NT2*32 = 768
    if (gid >= KT2 * NT2 * 32) return;
    int lane = gid & 31;
    int f    = gid >> 5;
    int nt   = f % NT2;
    int kt   = f / NT2;
    int n     = nt * 16 + (lane & 15);
    int kbase = kt * 32 + ((lane & 16) ? 8 : 0);
    _Float16* dst = out + (size_t)gid * 16;
#pragma unroll
    for (int r = 0; r < 8; ++r) {
#pragma unroll
        for (int p = 0; p < 2; ++p) {
            int k = kbase + 2 * r + p + ((r >= 4) ? 8 : 0); // always < 256
            float v = (n < NCLS) ? Wr[(size_t)k * NCLS + n] : 0.0f;
            dst[2 * r + p] = (_Float16)v;
        }
    }
}

// 2-round integer mix; one hash feeds TWO u16 uniforms (lo/hi halves).
__device__ __forceinline__ unsigned hash2(unsigned s)
{
    s *= 0x9E3779B9u; s ^= s >> 16;
    s *= 0x735A2D97u; s ^= s >> 15;
    return s;
}

union U16B { uint4 q; _Float16 h[8]; unsigned short us[8]; };

// ---------------------------------------------------------------------------
// Persistent fused SNN kernel: one block owns 64 batch rows for all 10 steps.
// v_hid / v_out live in VGPRs; hidden spikes cross between GEMMs via LDS.
// Wave w: layer-1 M-tile = w>>1, N-half = w&1 (8 N-tiles -> 8 WMMAs per A).
// ---------------------------------------------------------------------------
__global__ __launch_bounds__(256) void snn_fused_kernel(
    const unsigned short* __restrict__ thr,   // [4096, 800] u16 thresholds
    const _Float16* __restrict__ w1p,         // packed W1 B-fragments
    const _Float16* __restrict__ wrp,         // packed Wr B-fragments
    const float*    __restrict__ b1,          // [256]
    const float*    __restrict__ br,          // [47]
    float*          __restrict__ out)         // [4096, 47]
{
    __shared__ _Float16 spk[ROWS * HID];      // 32 KB hidden-spike tile

    const int tid  = threadIdx.x;
    const int lane = tid & 31;
    const int wid  = tid >> 5;                // 0..7
    const int mt   = wid >> 1;                // 0..3 : layer-1 M tile
    const int nh   = wid & 1;                 // 0..1 : layer-1 N half (8 tiles)
    const int l15  = lane & 15;
    const int hi   = (lane & 16) ? 1 : 0;
    const int rowBase = blockIdx.x * ROWS;
    const int arow    = rowBase + mt * 16 + l15;            // A-matrix row
    const unsigned short* trow = thr + (size_t)arow * KPAD;

    float b1v[8];
#pragma unroll
    for (int nt = 0; nt < 8; ++nt) b1v[nt] = b1[nh * 128 + nt * 16 + l15];
    float brv[NT2];
#pragma unroll
    for (int nt = 0; nt < NT2; ++nt) {
        int c = nt * 16 + l15;
        brv[nt] = (c < NCLS) ? br[c] : 0.0f;
    }

    v8f vh[8];                                // membrane state, C-layout
    v8f vout[NT2];                            // output EMA state
#pragma unroll
    for (int nt = 0; nt < 8; ++nt)
#pragma unroll
        for (int r = 0; r < 8; ++r) vh[nt][r] = 0.0f;
#pragma unroll
    for (int nt = 0; nt < NT2; ++nt)
#pragma unroll
        for (int r = 0; r < 8; ++r) vout[nt][r] = 0.0f;

    for (int t = 0; t < TSTEPS; ++t) {
        // ---------------- layer 1: I_hid = s_t @ W1 + b1 (WMMA f16) -----------
        v8f acc[8];                           // seeded with bias -> I = acc
#pragma unroll
        for (int nt = 0; nt < 8; ++nt)
#pragma unroll
            for (int r = 0; r < 8; ++r) acc[nt][r] = b1v[nt];

        const unsigned rbase = ((unsigned)t * (unsigned)BATCH + (unsigned)arow) * (unsigned)KPAD;
        for (int kt = 0; kt < KT1; ++kt) {
            // Bernoulli spikes built directly in 16-bit A-fragment layout:
            //   this lane holds K = kb..kb+7 and kb+16..kb+23 (two 16B runs).
            const int kb = kt * 32 + hi * 8;
            U16B tlo, thi2;
            tlo.q  = *(const uint4*)(trow + kb);            // thr K = kb..kb+7
            thi2.q = *(const uint4*)(trow + kb + 16);       // thr K = kb+16..kb+23
            v16h a;
#pragma unroll
            for (int p = 0; p < 8; ++p) {
                const unsigned k0 = (unsigned)kb + ((p < 4) ? 2 * p : 16 + 2 * (p - 4));
                const unsigned h  = hash2(rbase + k0);      // 2 uniforms per hash
                const unsigned short u0 = (unsigned short)h;
                const unsigned short u1 = (unsigned short)(h >> 16);
                const unsigned short t0 = (p < 4) ? tlo.us[2 * p]     : thi2.us[2 * (p - 4)];
                const unsigned short t1 = (p < 4) ? tlo.us[2 * p + 1] : thi2.us[2 * (p - 4) + 1];
                a[2 * p]     = (u0 < t0) ? (_Float16)1.0f : (_Float16)0.0f;
                a[2 * p + 1] = (u1 < t1) ? (_Float16)1.0f : (_Float16)0.0f;
            }
            const _Float16* wb = w1p + ((size_t)((kt * NT1 + nh * 8) * 32 + lane)) * 16;
#pragma unroll
            for (int nt = 0; nt < 8; ++nt) {
                v16h b = *(const v16h*)(wb + (size_t)nt * 32 * 16);  // 2x b128 loads
                acc[nt] = __builtin_amdgcn_wmma_f32_16x16x32_f16(
                    false, a, false, b, (short)0, acc[nt], false, false);
            }
        }

        // ---- LIF: v += (I-v)/tau, spike select, hard reset; spikes -> LDS ----
#pragma unroll
        for (int nt = 0; nt < 8; ++nt) {
#pragma unroll
            for (int r = 0; r < 8; ++r) {
                const float v = 0.5f * (vh[nt][r] + acc[nt][r]);
                const bool fire = (v >= 1.0f);               // Heaviside(v-1)
                vh[nt][r] = fire ? 0.0f : v;                 // hard reset (select)
                const _Float16 sh = fire ? (_Float16)1.0f : (_Float16)0.0f;
                const int row = mt * 16 + r + hi * 8;        // C-layout row
                const int col = nh * 128 + nt * 16 + l15;    // C-layout col
                spk[row * HID + col] = sh;
            }
        }
        __syncthreads();

        // ---------- layer 2: I = spk @ Wr + br (waves 0..3, uniform) ----------
        if (wid < 4) {
            v8f acc2[NT2];
#pragma unroll
            for (int nt = 0; nt < NT2; ++nt)
#pragma unroll
                for (int r = 0; r < 8; ++r) acc2[nt][r] = brv[nt];
            const int row2 = wid * 16 + l15;
#pragma unroll
            for (int kt = 0; kt < KT2; ++kt) {
                const int kb = kt * 32 + hi * 8;
                U16B lo, hi8;
                lo.q  = *(const uint4*)(&spk[row2 * HID + kb]);      // ds_load_b128
                hi8.q = *(const uint4*)(&spk[row2 * HID + kb + 16]); // ds_load_b128
                v16h a2;
#pragma unroll
                for (int p = 0; p < 8; ++p) { a2[p] = lo.h[p]; a2[8 + p] = hi8.h[p]; }
                const _Float16* wb = wrp + ((size_t)((kt * NT2) * 32 + lane)) * 16;
#pragma unroll
                for (int nt = 0; nt < NT2; ++nt) {
                    v16h b = *(const v16h*)(wb + (size_t)nt * 32 * 16);
                    acc2[nt] = __builtin_amdgcn_wmma_f32_16x16x32_f16(
                        false, a2, false, b, (short)0, acc2[nt], false, false);
                }
            }
#pragma unroll
            for (int nt = 0; nt < NT2; ++nt)
#pragma unroll
                for (int r = 0; r < 8; ++r)
                    vout[nt][r] = 0.5f * (vout[nt][r] + acc2[nt][r]);  // EMA
        }
        __syncthreads();
    }

    // ------------------------------ write v_out ------------------------------
    if (wid < 4) {
#pragma unroll
        for (int nt = 0; nt < NT2; ++nt) {
#pragma unroll
            for (int r = 0; r < 8; ++r) {
                int row = rowBase + wid * 16 + r + hi * 8;
                int col = nt * 16 + l15;
                if (col < NCLS) out[(size_t)row * NCLS + col] = vout[nt][r];
            }
        }
    }
}

extern "C" void kernel_launch(void* const* d_in, const int* in_sizes, int n_in,
                              void* d_out, int out_size, void* d_ws, size_t ws_size,
                              hipStream_t stream)
{
    (void)in_sizes; (void)n_in; (void)out_size; (void)ws_size;
    const float* x  = (const float*)d_in[0];   // [4096,1,28,28]
    const float* W1 = (const float*)d_in[1];   // [784,256]
    const float* b1 = (const float*)d_in[2];   // [256]
    const float* Wr = (const float*)d_in[3];   // [256,47]
    const float* br = (const float*)d_in[4];   // [47]
    float* out = (float*)d_out;                // [4096,47]

    unsigned short* thr = (unsigned short*)d_ws;                 // 6,553,600 B
    _Float16* w1p = (_Float16*)((char*)d_ws + (size_t)BATCH * KPAD * 2);
    _Float16* wrp = (_Float16*)((char*)w1p + (size_t)KT1 * NT1 * 32 * 16 * 2);

    pack_thr_kernel<<<(BATCH * KPAD + 255) / 256, 256, 0, stream>>>(x, thr);
    pack_w1_kernel<<<(KT1 * NT1 * 32 + 255) / 256, 256, 0, stream>>>(W1, w1p);
    pack_wr_kernel<<<(KT2 * NT2 * 32 + 255) / 256, 256, 0, stream>>>(Wr, wrp);
    snn_fused_kernel<<<NBLK, 256, 0, stream>>>(thr, w1p, wrp, b1, br, out);
}
